// MMAgentCrossAttention_53626961658248
// MI455X (gfx1250) — matrix operand order repaired and checked by reading, not compiled, not run on hardware
//
#include <hip/hip_runtime.h>
#include <hip/hip_bf16.h>
#include <stdint.h>

// ---------------------------------------------------------------------------
// MI455X (gfx1250) agent-attention.
//  - all matmuls on v_wmma_f32_16x16x32_bf16 (fp32 accumulate)
//  - operands pre-converted to bf16; weights pre-transposed to NxK
//  - dominant GEMMs (qkv, proj): 128x128 block tile, 32x64 wave tile,
//    double-buffered global_load_async_to_lds_b128 with a branch-free
//    steady-state loop: s_wait_asynccnt -> ds_load_b128 -> 8x v_wmma
// ---------------------------------------------------------------------------

typedef unsigned short u16;
typedef __attribute__((ext_vector_type(16))) __bf16 v16bf;
typedef __attribute__((ext_vector_type(8)))  float  v8f;

#define GEMM_THREADS 256
#define BLK_K 32
#define LDSTR 40   // bf16 elems per LDS row: 80B stride (16B aligned, bank-skewed)

// guarded-kernel tile
#define GBLK_M 128
#define GBLK_N 64
// fast-kernel tile
#define FBLK_M 128
#define FBLK_N 128

__device__ __forceinline__ u16 f2bf(float f) {
  union { float f; unsigned int u; } c; c.f = f;
  unsigned int u = c.u;
  unsigned int r = u + 0x7FFFu + ((u >> 16) & 1u);   // round-to-nearest-even
  return (u16)(r >> 16);
}
__device__ __forceinline__ float bf2f(u16 h) {
  union { unsigned int u; float f; } c; c.u = ((unsigned int)h) << 16;
  return c.f;
}

union FragU {
  uint4  q[2];
  v16bf  v;
};

// low 32 bits of a flat pointer to LDS == LDS byte offset
__device__ __forceinline__ unsigned lds_off32(const void* p) {
  return (unsigned)(unsigned long long)p;
}
__device__ __forceinline__ void async_g2l_b128(unsigned ldsOff, const void* gaddr) {
  asm volatile("global_load_async_to_lds_b128 %0, %1, off"
               :: "v"(ldsOff), "v"(gaddr) : "memory");
}

// ======================= FAST GEMM (exact-tile dims) ========================
// C = alpha * A @ B^T + bias.  A: MxK bf16 (lda), Bt: NxK bf16 (ldb).
// Requires M%128==0, N%128==0, K%32==0, lda%8==0, ldb%8==0.
template<bool OUTBF16>
__global__ __launch_bounds__(GEMM_THREADS)
void gemm_fast(const u16* __restrict__ A, int lda,
               const u16* __restrict__ Bt, int ldb,
               void* __restrict__ Cv, int ldc,
               const float* __restrict__ bias,
               int K, float alpha)
{
  __shared__ __align__(16) u16 shA[2][FBLK_M * LDSTR];
  __shared__ __align__(16) u16 shB[2][FBLK_N * LDSTR];

  const int tid  = threadIdx.x;
  const int lane = tid & 31;
  const int wave = tid >> 5;
  const int wm   = wave & 3;    // 4 row groups of 32
  const int wn   = wave >> 2;   // 2 col groups of 64

  const int rowBase = blockIdx.y * FBLK_M;
  const int colBase = blockIdx.x * FBLK_N;

  const v8f vzero = {0.f,0.f,0.f,0.f,0.f,0.f,0.f,0.f};
  v8f acc[2][4] = {{vzero, vzero, vzero, vzero}, {vzero, vzero, vzero, vzero}};

  // copy mapping (identical for A and B tiles: 128 rows x 32 bf16)
  const int cRow = tid & 127;
  const int cKh  = (tid >> 7) << 4;     // 0 / 16

  unsigned ldsA[2], ldsB[2];
  ldsA[0] = lds_off32(&shA[0][cRow * LDSTR + cKh]);
  ldsA[1] = lds_off32(&shA[1][cRow * LDSTR + cKh]);
  ldsB[0] = lds_off32(&shB[0][cRow * LDSTR + cKh]);
  ldsB[1] = lds_off32(&shB[1][cRow * LDSTR + cKh]);

  const u16* gA = A + (long)(rowBase + cRow) * lda + cKh;
  const u16* gB = Bt + (long)(colBase + cRow) * ldb + cKh;

  // async copy of tile at k0 into buffer buf (4 async ops per wave)
  auto issue = [&](int k0, int buf) {
    async_g2l_b128(ldsA[buf],      gA + k0);
    async_g2l_b128(ldsA[buf] + 16, gA + k0 + 8);
    async_g2l_b128(ldsB[buf],      gB + k0);
    async_g2l_b128(ldsB[buf] + 16, gB + k0 + 8);
  };

  const int half = lane >> 4;
  const int r15  = lane & 15;

  auto compute = [&](int buf) {
    FragU aF[2], bF[4];
    #pragma unroll
    for (int i = 0; i < 2; i++) {
      const u16* p = &shA[buf][(wm * 32 + i * 16 + r15) * LDSTR + half * 8];
      aF[i].q[0] = *(const uint4*)p;
      aF[i].q[1] = *(const uint4*)(p + 16);
    }
    #pragma unroll
    for (int j = 0; j < 4; j++) {
      const u16* p = &shB[buf][(wn * 64 + j * 16 + r15) * LDSTR + half * 8];
      bF[j].q[0] = *(const uint4*)p;
      bF[j].q[1] = *(const uint4*)(p + 16);
    }
    #pragma unroll
    for (int i = 0; i < 2; i++)
      #pragma unroll
      for (int j = 0; j < 4; j++)
        acc[i][j] = __builtin_amdgcn_wmma_f32_16x16x32_bf16(
            false, aF[i].v, false, bF[j].v, (short)0, acc[i][j], false, false);
  };

  const int nk = K / BLK_K;
  issue(0, 0);
  if (nk > 1) {
    issue(BLK_K, 1);
    // steady state: branch-free body
    for (int t = 0; t < nk - 2; ++t) {
      asm volatile("s_wait_asynccnt 0x4" ::: "memory");
      __syncthreads();
      compute(t & 1);
      __syncthreads();
      issue((t + 2) * BLK_K, t & 1);
    }
    // tile nk-2 (one tile still in flight)
    asm volatile("s_wait_asynccnt 0x4" ::: "memory");
    __syncthreads();
    compute(nk & 1);
    __syncthreads();
  }
  // last tile
  asm volatile("s_wait_asynccnt 0x0" ::: "memory");
  __syncthreads();
  compute((nk - 1) & 1);

  const int hi8 = (lane >> 4) * 8;
  #pragma unroll
  for (int i = 0; i < 2; i++)
    #pragma unroll
    for (int j = 0; j < 4; j++) {
      const int col = colBase + wn * 64 + j * 16 + r15;
      const float bv = bias ? bias[col] : 0.f;
      const long row0 = rowBase + wm * 32 + i * 16 + hi8;
      #pragma unroll
      for (int r = 0; r < 8; r++) {
        const float val = acc[i][j][r] * alpha + bv;
        const long  idx = (row0 + r) * (long)ldc + col;
        if (OUTBF16) ((u16*)Cv)[idx] = f2bf(val);
        else         ((float*)Cv)[idx] = val;
      }
    }
}

// ======================= GUARDED GEMM (odd shapes, batched) =================
// C[z] = alpha * A[z] @ op(B[z]) + bias ; op = transpose if TRANSB (B: NxK)
template<bool TRANSB, bool OUTBF16>
__global__ __launch_bounds__(GEMM_THREADS)
void gemm_guard(const u16* __restrict__ A, int lda, long sAb, long sAh,
                const u16* __restrict__ B, int ldb, long sBb, long sBh,
                void* __restrict__ Cv, int ldc, long sCb, long sCh,
                const float* __restrict__ bias,
                int M, int N, int K, float alpha, int ZH)
{
  __shared__ __align__(16) u16 shA[GBLK_M * LDSTR];
  __shared__ __align__(16) u16 shB[GBLK_N * LDSTR];

  const int tid  = threadIdx.x;
  const int lane = tid & 31;
  const int wave = tid >> 5;
  const int wm   = wave & 3;
  const int wn   = wave >> 2;

  const int zb = blockIdx.z / ZH;
  const int zh = blockIdx.z % ZH;
  A += (long)zb * sAb + (long)zh * sAh;
  B += (long)zb * sBb + (long)zh * sBh;
  char* C = (char*)Cv + ((long)zb * sCb + (long)zh * sCh) * (OUTBF16 ? 2 : 4);

  const int rowBase = blockIdx.y * GBLK_M;
  const int colBase = blockIdx.x * GBLK_N;

  const v8f vzero = {0.f,0.f,0.f,0.f,0.f,0.f,0.f,0.f};
  v8f acc[2][2] = {{vzero, vzero}, {vzero, vzero}};

  const int aRow = tid & 127;
  const int aKh  = (tid >> 7) << 4;
  const int bK   = tid & 31;            // non-trans
  const int bNc  = (tid >> 5) << 3;
  const int tN   = tid & 63;            // trans
  const int tKq  = (tid >> 6) << 3;

  const bool aVec = ((lda & 7) == 0);
  const bool bVec = ((ldb & 7) == 0);

  const int half = lane >> 4;
  const int r15  = lane & 15;

  for (int k0 = 0; k0 < K; k0 += BLK_K) {
    __syncthreads();
    { // A tile
      const long grow = rowBase + aRow;
      const u16* ap = A + grow * (long)lda + k0 + aKh;
      u16* dst = &shA[aRow * LDSTR + aKh];
      if (grow < M && (k0 + aKh + 15) < K && aVec) {
        uint4 q0 = ((const uint4*)ap)[0];
        uint4 q1 = ((const uint4*)ap)[1];
        ((uint4*)dst)[0] = q0;
        ((uint4*)dst)[1] = q1;
      } else {
        #pragma unroll
        for (int i = 0; i < 16; i++)
          dst[i] = (grow < M && (k0 + aKh + i) < K) ? ap[i] : (u16)0;
      }
    }
    if (TRANSB) {
      const u16* bp = B + (long)(colBase + tN) * ldb + k0 + tKq;
      u16* dst = &shB[tN * LDSTR + tKq];
      if ((colBase + tN) < N && (k0 + tKq + 7) < K && bVec) {
        ((uint4*)dst)[0] = *(const uint4*)bp;
      } else {
        #pragma unroll
        for (int i = 0; i < 8; i++)
          dst[i] = ((colBase + tN) < N && (k0 + tKq + i) < K) ? bp[i] : (u16)0;
      }
    } else {
      const u16* bp = B + (long)(k0 + bK) * ldb + colBase + bNc;
      #pragma unroll
      for (int j = 0; j < 8; j++)
        shB[(bNc + j) * LDSTR + bK] =
            ((k0 + bK) < K && (colBase + bNc + j) < N) ? bp[j] : (u16)0;
    }
    __syncthreads();

    FragU aF[2], bF[2];
    #pragma unroll
    for (int i = 0; i < 2; i++) {
      const u16* p = &shA[(wm * 32 + i * 16 + r15) * LDSTR + half * 8];
      aF[i].q[0] = *(const uint4*)p;
      aF[i].q[1] = *(const uint4*)(p + 16);
    }
    #pragma unroll
    for (int j = 0; j < 2; j++) {
      const u16* p = &shB[(wn * 32 + j * 16 + r15) * LDSTR + half * 8];
      bF[j].q[0] = *(const uint4*)p;
      bF[j].q[1] = *(const uint4*)(p + 16);
    }
    #pragma unroll
    for (int i = 0; i < 2; i++)
      #pragma unroll
      for (int j = 0; j < 2; j++)
        acc[i][j] = __builtin_amdgcn_wmma_f32_16x16x32_bf16(
            false, aF[i].v, false, bF[j].v, (short)0, acc[i][j], false, false);
  }

  const int hi8 = (lane >> 4) * 8;
  #pragma unroll
  for (int i = 0; i < 2; i++)
    #pragma unroll
    for (int j = 0; j < 2; j++) {
      const int col = colBase + wn * 32 + j * 16 + r15;
      if (col < N) {
        const float bv = bias ? bias[col] : 0.f;
        const long row0 = rowBase + wm * 32 + i * 16 + hi8;
        #pragma unroll
        for (int r = 0; r < 8; r++) {
          const long row = row0 + r;
          if (row < M) {
            const float val = acc[i][j][r] * alpha + bv;
            if (OUTBF16) ((u16*)C)[row * (long)ldc + col] = f2bf(val);
            else         ((float*)C)[row * (long)ldc + col] = val;
          }
        }
      }
    }
}

// ============================== helpers =====================================

// elementwise fp32 -> bf16, 8 elems/thread (count % 2048 == 0)
__global__ void cvt_bf16(const float* __restrict__ src, u16* __restrict__ dst) {
  long i8 = ((long)blockIdx.x * 256 + threadIdx.x) * 8;
  float4 f0 = ((const float4*)(src + i8))[0];
  float4 f1 = ((const float4*)(src + i8))[1];
  __align__(16) u16 h[8] = { f2bf(f0.x), f2bf(f0.y), f2bf(f0.z), f2bf(f0.w),
                             f2bf(f1.x), f2bf(f1.y), f2bf(f1.z), f2bf(f1.w) };
  *(uint4*)(dst + i8) = *(const uint4*)h;
}

// src: R x C fp32 -> dst: C x R bf16 (transpose + convert)
__global__ void cvt_transpose(const float* __restrict__ src, u16* __restrict__ dst,
                              int R, int C) {
  __shared__ u16 t[32][33];
  const int c0 = blockIdx.x * 32, r0 = blockIdx.y * 32;
  const int tx = threadIdx.x, ty = threadIdx.y;
  for (int i = ty; i < 32; i += 8) {
    int r = r0 + i, c = c0 + tx;
    t[i][tx] = (r < R && c < C) ? f2bf(src[(long)r * C + c]) : (u16)0;
  }
  __syncthreads();
  for (int i = ty; i < 32; i += 8) {
    int c = c0 + i, r = r0 + tx;
    if (c < C && r < R) dst[(long)c * R + r] = t[tx][i];
  }
}

// xt = concat(x_t, broadcast tokens): (8,37,4096) in bf16
__global__ void xt_build(const float* __restrict__ x_t, const float* __restrict__ tokens,
                         u16* __restrict__ xt) {
  long idx = (long)blockIdx.x * 256 + threadIdx.x;
  if (idx >= (long)8 * 37 * 4096) return;
  int d = idx & 4095;
  long rt = idx >> 12;
  int t = (int)(rt % 37);
  int b = (int)(rt / 37);
  float v = (t < 32) ? x_t[((long)b * 32 + t) * 4096 + d]
                     : tokens[(long)(t - 32) * 4096 + d];
  xt[idx] = f2bf(v);
}

// softmax along 4096 rows: fp32 logits in, bf16 probs out
__global__ void softmax_rows_4096(const float* __restrict__ S, u16* __restrict__ P) {
  const float* row = S + (long)blockIdx.x * 4096;
  u16* prow = P + (long)blockIdx.x * 4096;
  __shared__ float red[256];
  int tid = threadIdx.x;
  float m = -3.4e38f;
  for (int i = tid; i < 4096; i += 256) m = fmaxf(m, row[i]);
  red[tid] = m; __syncthreads();
  for (int s = 128; s > 0; s >>= 1) { if (tid < s) red[tid] = fmaxf(red[tid], red[tid + s]); __syncthreads(); }
  m = red[0]; __syncthreads();
  float sum = 0.f;
  for (int i = tid; i < 4096; i += 256) sum += __expf(row[i] - m);
  red[tid] = sum; __syncthreads();
  for (int s = 128; s > 0; s >>= 1) { if (tid < s) red[tid] += red[tid + s]; __syncthreads(); }
  float inv = 1.f / red[0];
  for (int i = tid; i < 4096; i += 256) prow[i] = f2bf(__expf(row[i] - m) * inv);
}

// softmax along 37 rows: fp32 in, bf16 out; one wave32 per row
__global__ void softmax_rows_37(const float* __restrict__ T, u16* __restrict__ P) {
  long row = (long)blockIdx.x * 8 + (threadIdx.x >> 5);
  int lane = threadIdx.x & 31;
  const float* p = T + row * 37;
  u16* q = P + row * 37;
  float v0 = p[lane];
  float v1 = (lane + 32 < 37) ? p[lane + 32] : -3.4e38f;
  float m = fmaxf(v0, v1);
  for (int o = 16; o; o >>= 1) m = fmaxf(m, __shfl_xor(m, o));
  float e0 = __expf(v0 - m);
  float e1 = (lane + 32 < 37) ? __expf(v1 - m) : 0.f;
  float s = e0 + e1;
  for (int o = 16; o; o >>= 1) s += __shfl_xor(s, o);
  float inv = 1.f / s;
  q[lane] = f2bf(e0 * inv);
  if (lane + 32 < 37) q[lane + 32] = f2bf(e1 * inv);
}

// y_bf = bf16(out_pre + depthwise conv(v)) ; v read from bf16 qkv slice
__global__ void conv_add_cvt(const u16* __restrict__ qkv_bf, const float* __restrict__ dwc_w,
                             const float* __restrict__ dwc_b, const float* __restrict__ out_pre,
                             u16* __restrict__ y_bf) {
  long idx = (long)blockIdx.x * 256 + threadIdx.x;   // < 8*4096*1024
  int c = (int)(idx & 1023);
  int n = (int)((idx >> 10) & 4095);
  int b = (int)(idx >> 22);
  const long base = (long)b * 4096 * 3072 + 2048 + c;
  float vm = (n > 0)    ? bf2f(qkv_bf[base + (long)(n - 1) * 3072]) : 0.f;
  float vc =              bf2f(qkv_bf[base + (long)n * 3072]);
  float vp = (n < 4095) ? bf2f(qkv_bf[base + (long)(n + 1) * 3072]) : 0.f;
  float conv = vm * dwc_w[c * 3 + 0] + vc * dwc_w[c * 3 + 1] + vp * dwc_w[c * 3 + 2] + dwc_b[c];
  y_bf[idx] = f2bf(out_pre[idx] + conv);
}

// ---------------------------------------------------------------------------

extern "C" void kernel_launch(void* const* d_in, const int* in_sizes, int n_in,
                              void* d_out, int out_size, void* d_ws, size_t ws_size,
                              hipStream_t stream) {
  const float* x      = (const float*)d_in[0];
  const float* x_t    = (const float*)d_in[1];
  const float* tokens = (const float*)d_in[2];
  const float* W_qkv  = (const float*)d_in[3];  // (1024,3072)
  const float* W_agent= (const float*)d_in[4];  // (4096,1024)
  const float* W_proj = (const float*)d_in[5];  // (1024,1024)
  const float* b_proj = (const float*)d_in[6];
  const float* dwc_w  = (const float*)d_in[7];
  const float* dwc_b  = (const float*)d_in[8];
  float* out = (float*)d_out;

  const float scale = 0.125f;

  // workspace layout (byte offsets, all 256B-aligned)
  char* w = (char*)d_ws;
  u16*   qkv_bf = (u16*)w;               w += 201326592;  // 32768x3072 bf16
  u16*   x_bf   = (u16*)w;               w += 67108864;   // 32768x1024 bf16
  u16*   wqkvT  = (u16*)w;               w += 6291456;    // 3072x1024 bf16
  u16*   wagT   = (u16*)w;               w += 8388608;    // 1024x4096 bf16
  u16*   wprojT = (u16*)w;               w += 2097152;    // 1024x1024 bf16
  u16*   xt_bf  = (u16*)w;               w += 2424832;    // 296x4096 bf16
  u16*   a_bf   = (u16*)w;               w += 606208;     // 296x1024 bf16
  float* Sf     = (float*)w;             w += 77594624;   // 128x37x4096 f32 (also T)
  u16*   Pp     = (u16*)w;               w += 38797312;   // 128x37x4096 bf16 (also Tp)
  u16*   av_bf  = (u16*)w;               w += 606208;     // 128x37x64 bf16
  float* out_pre= (float*)w;             w += 134217728;  // 32768x1024 f32
  u16*   y_bf   = (u16*)w;               w += 67108864;   // 32768x1024 bf16

  // --- operand conversion ---
  cvt_bf16<<<dim3(16384), dim3(256), 0, stream>>>(x, x_bf);              // 33554432
  cvt_transpose<<<dim3(96, 32), dim3(32, 8), 0, stream>>>(W_qkv, wqkvT, 1024, 3072);
  cvt_transpose<<<dim3(32, 128), dim3(32, 8), 0, stream>>>(W_agent, wagT, 4096, 1024);
  cvt_transpose<<<dim3(32, 32), dim3(32, 8), 0, stream>>>(W_proj, wprojT, 1024, 1024);
  xt_build<<<dim3(4736), dim3(256), 0, stream>>>(x_t, tokens, xt_bf);

  // --- qkv = x @ W_qkv (FAST, async, bf16 out): M=32768 N=3072 K=1024 ---
  gemm_fast<true><<<dim3(24, 256), dim3(GEMM_THREADS), 0, stream>>>(
      x_bf, 1024, wqkvT, 1024, qkv_bf, 3072, nullptr, 1024, 1.f);

  // --- a = xt @ W_agent : M=296 N=1024 K=4096 ---
  gemm_guard<true, true><<<dim3(16, 3, 1), dim3(GEMM_THREADS), 0, stream>>>(
      xt_bf, 4096, 0, 0, wagT, 4096, 0, 0, a_bf, 1024, 0, 0,
      nullptr, 296, 1024, 4096, 1.f, 1);

  // --- S[bh] = scale * a @ k^T : M=37 N=4096 K=64, batch 128 ---
  gemm_guard<true, false><<<dim3(64, 1, 128), dim3(GEMM_THREADS), 0, stream>>>(
      a_bf, 1024, (long)37 * 1024, 64,
      qkv_bf + 1024, 3072, (long)4096 * 3072, 64,
      Sf, 4096, (long)16 * 37 * 4096, (long)37 * 4096,
      nullptr, 37, 4096, 64, scale, 16);

  softmax_rows_4096<<<dim3(128 * 37), dim3(256), 0, stream>>>(Sf, Pp);

  // --- agent_v[bh] = attn @ v : M=37 N=64 K=4096 ---
  gemm_guard<false, true><<<dim3(1, 1, 128), dim3(GEMM_THREADS), 0, stream>>>(
      Pp, 4096, (long)16 * 37 * 4096, (long)37 * 4096,
      qkv_bf + 2048, 3072, (long)4096 * 3072, 64,
      av_bf, 64, (long)16 * 37 * 64, (long)37 * 64,
      nullptr, 37, 64, 4096, 1.f, 16);

  // --- T[bh] = scale * q @ a^T : M=4096 N=37 K=64 (T aliases Sf) ---
  gemm_guard<true, false><<<dim3(1, 32, 128), dim3(GEMM_THREADS), 0, stream>>>(
      qkv_bf, 3072, (long)4096 * 3072, 64,
      a_bf, 1024, (long)37 * 1024, 64,
      Sf, 37, (long)16 * 4096 * 37, (long)4096 * 37,
      nullptr, 4096, 37, 64, scale, 16);

  softmax_rows_37<<<dim3(65536), dim3(256), 0, stream>>>(Sf, Pp);

  // --- out_pre[bh] = q_attn @ agent_v : M=4096 N=64 K=37 ---
  gemm_guard<false, false><<<dim3(1, 32, 128), dim3(GEMM_THREADS), 0, stream>>>(
      Pp, 37, (long)16 * 4096 * 37, (long)4096 * 37,
      av_bf, 64, (long)16 * 37 * 64, (long)37 * 64,
      out_pre, 1024, (long)4096 * 1024, 64,
      nullptr, 4096, 64, 37, 1.f, 16);

  // --- y = bf16(out_pre + depthwise conv residual) ---
  conv_add_cvt<<<dim3(131072), dim3(256), 0, stream>>>(qkv_bf, dwc_w, dwc_b, out_pre, y_bf);

  // --- out = y @ W_proj + b_proj (FAST, async, fp32 out): N=1024 K=1024 ---
  gemm_fast<false><<<dim3(8, 256), dim3(GEMM_THREADS), 0, stream>>>(
      y_bf, 1024, wprojT, 1024, out, 1024, b_proj, 1024, 1.f);

  (void)in_sizes; (void)n_in; (void)out_size; (void)ws_size;
}
